// NeuralLongTermMemory_46806553591932
// MI455X (gfx1250) — compile-verified, matrix operand order checked by reading, not run on hardware
//
#include <hip/hip_runtime.h>
#include <hip/hip_bf16.h>

// ---------------------------------------------------------------------------
// Titans-style neural long-term memory for MI455X (gfx1250), wave32 + WMMA.
// All einsums run through v_wmma_f32_16x16x32_f16 (f16 operands, f32 accum).
// ---------------------------------------------------------------------------

typedef __attribute__((ext_vector_type(16))) _Float16 v16h;
typedef __attribute__((ext_vector_type(4)))  _Float16 v4h;
typedef __attribute__((ext_vector_type(8)))  float    v8f;

#define DIMN   1024
#define HIDN   2048
#define BSZ    4
#define SEQ    2048
#define CHUNKN 64
#define NCH    (SEQ / CHUNKN)        // 32
#define TOKENS (BSZ * SEQ)           // 8192
#define CROWS  (BSZ * CHUNKN)        // 256 rows per chunk

#define TM 128
#define TN 128
#define TK 32
#define LDSK 40                       // padded LDS stride (halfs): 80B rows

enum { EPI_NONE = 0, EPI_SILU = 1, EPI_SILU2 = 2, EPI_D2 = 3, EPI_D1 = 4, EPI_UPDATE = 5 };

// v_rcp_f32 instead of correctly-rounded divide: kills the v_div_scale/
// div_fixup chains the compiler otherwise emits for 1/(1+e^-x).
__device__ __forceinline__ float frcp_(float x)     { return __builtin_amdgcn_rcpf(x); }
__device__ __forceinline__ float sigmoidf_(float x) { return frcp_(1.0f + __expf(-x)); }
__device__ __forceinline__ float siluf_(float x)    { return x * sigmoidf_(x); }
__device__ __forceinline__ float silu_d_(float x)   { float s = sigmoidf_(x); return s * (1.0f + x * (1.0f - s)); }
__device__ __forceinline__ float clip1_(float x)    { return fminf(fmaxf(x, -1.0f), 1.0f); }

// token-linear index (b*SEQ + c*CHUNK + j) -> chunk-permuted row ((c*B + b)*CHUNK + j)
__device__ __forceinline__ int permute_token(int m) {
    int b = m >> 11;          // / SEQ
    int s = m & (SEQ - 1);
    int c = s >> 6;           // / CHUNK
    int j = s & (CHUNKN - 1);
    return ((c << 2) + b) * CHUNKN + j;
}

union FragH { v16h v; uint4 q[2]; };

// A-fragment layout (16-bit, 16x32): lane L holds row M = L%16; K chunks
// [kb..kb+7] and [kb+16..kb+23] with kb = (L/16)*8 -> two aligned b128 reads.
// B fragment uses the mirrored layout (lane = N, same K chunks), i.e. B loaded
// "column major" which matches our Bs tile = B^T staging.
__device__ __forceinline__ v16h frag_from_lds(const _Float16* tile, int row, int kb) {
    FragH f;
    const _Float16* p = tile + row * LDSK + kb;
    f.q[0] = *reinterpret_cast<const uint4*>(p);
    f.q[1] = *reinterpret_cast<const uint4*>(p + 16);
    return f.v;
}

// AT: 0 = A stored [M,K]; 1 = A stored [K,M]; 2 = A stored [M,K] with gathered
//     (chunk-permuted) rows.  BT: 0 = B stored [N,K]; 1 = B stored [K,N].
template <int AT, int BT, int EPI, bool PERM_OUT>
__global__ __launch_bounds__(256) void wmma_gemm(
    const float* __restrict__ A, const float* __restrict__ Bm,
    float* __restrict__ C, float* __restrict__ C2,
    const float* __restrict__ aux,
    float* __restrict__ Wbuf, float* __restrict__ Mbuf,
    const float* __restrict__ gates, int gidx,
    int M, int N, int K, float scaleArg)
{
    __shared__ __align__(16) _Float16 As[TM * LDSK];
    __shared__ __align__(16) _Float16 Bs[TN * LDSK];

    const int t    = threadIdx.x;
    const int lane = t & 31;
    const int wave = t >> 5;
    const int m0   = blockIdx.y * TM;
    const int n0   = blockIdx.x * TN;
    const int wm   = wave & 3;   // 4 wave-rows  -> 32 rows each
    const int wn   = wave >> 2;  // 2 wave-cols  -> 64 cols each

    v8f acc[2][4] = {};

    for (int k0 = 0; k0 < K; k0 += TK) {
        // ---- stage A tile: As[r][kk] = A(m0+r, k0+kk) as f16 -------------
        // Next-tile prefetches are unconditional: prefetch is speculative and
        // never faults the wave (OOB translations are silently dropped).
        if constexpr (AT == 0 || AT == 2) {
            #pragma unroll
            for (int i = 0; i < 4; ++i) {
                int qid = i * 256 + t;
                int row = qid >> 3;
                int c4  = (qid & 7) << 2;
                int gr  = m0 + row;
                if constexpr (AT == 2) gr = permute_token(gr);
                const float* src = A + (size_t)gr * K + k0 + c4;
                float4 f = *reinterpret_cast<const float4*>(src);
                __builtin_prefetch(src + TK, 0, 3);  // global_prefetch_b8 (near)
                v4h h; h.x = (_Float16)f.x; h.y = (_Float16)f.y;
                       h.z = (_Float16)f.z; h.w = (_Float16)f.w;
                *reinterpret_cast<v4h*>(&As[row * LDSK + c4]) = h;
            }
        } else {  // AT == 1: contiguous along M in memory
            #pragma unroll
            for (int i = 0; i < 4; ++i) {
                int qid = i * 256 + t;
                int kk  = qid >> 5;
                int r4  = (qid & 31) << 2;
                const float* src = A + (size_t)(k0 + kk) * M + m0 + r4;
                float4 f = *reinterpret_cast<const float4*>(src);
                __builtin_prefetch(src + (size_t)TK * M, 0, 3);
                As[(r4 + 0) * LDSK + kk] = (_Float16)f.x;
                As[(r4 + 1) * LDSK + kk] = (_Float16)f.y;
                As[(r4 + 2) * LDSK + kk] = (_Float16)f.z;
                As[(r4 + 3) * LDSK + kk] = (_Float16)f.w;
            }
        }
        // ---- stage B tile: Bs[j][kk] = B(k0+kk, n0+j) as f16 -------------
        if constexpr (BT == 0) {  // B stored [N,K], contiguous along K
            #pragma unroll
            for (int i = 0; i < 4; ++i) {
                int qid = i * 256 + t;
                int row = qid >> 3;
                int c4  = (qid & 7) << 2;
                const float* src = Bm + (size_t)(n0 + row) * K + k0 + c4;
                float4 f = *reinterpret_cast<const float4*>(src);
                __builtin_prefetch(src + TK, 0, 3);
                v4h h; h.x = (_Float16)f.x; h.y = (_Float16)f.y;
                       h.z = (_Float16)f.z; h.w = (_Float16)f.w;
                *reinterpret_cast<v4h*>(&Bs[row * LDSK + c4]) = h;
            }
        } else {                  // B stored [K,N], contiguous along N
            #pragma unroll
            for (int i = 0; i < 4; ++i) {
                int qid = i * 256 + t;
                int kk  = qid >> 5;
                int r4  = (qid & 31) << 2;
                const float* src = Bm + (size_t)(k0 + kk) * N + n0 + r4;
                float4 f = *reinterpret_cast<const float4*>(src);
                __builtin_prefetch(src + (size_t)TK * N, 0, 3);
                Bs[(r4 + 0) * LDSK + kk] = (_Float16)f.x;
                Bs[(r4 + 1) * LDSK + kk] = (_Float16)f.y;
                Bs[(r4 + 2) * LDSK + kk] = (_Float16)f.z;
                Bs[(r4 + 3) * LDSK + kk] = (_Float16)f.w;
            }
        }
        __syncthreads();

        const int fr = lane & 15;
        const int kb = (lane >> 4) * 8;
        v16h af[2], bf[4];
        #pragma unroll
        for (int mi = 0; mi < 2; ++mi) af[mi] = frag_from_lds(As, wm * 32 + mi * 16 + fr, kb);
        #pragma unroll
        for (int nj = 0; nj < 4; ++nj) bf[nj] = frag_from_lds(Bs, wn * 64 + nj * 16 + fr, kb);

        #pragma unroll
        for (int mi = 0; mi < 2; ++mi)
            #pragma unroll
            for (int nj = 0; nj < 4; ++nj)
                acc[mi][nj] = __builtin_amdgcn_wmma_f32_16x16x32_f16(
                    false, af[mi], false, bf[nj], (short)0, acc[mi][nj], false, false);
        __syncthreads();
    }

    // ---- epilogue: C layout lane = N, (vgpr, lane/16) = M -----------------
    const int lcol = lane & 15;
    const int lrow = (lane >> 4) * 8;
    float g_lr = 0.f, g_al = 0.f, g_et = 0.f;
    if constexpr (EPI == EPI_UPDATE) {
        g_lr = gates[gidx * 3 + 0];
        g_al = gates[gidx * 3 + 1];
        g_et = gates[gidx * 3 + 2];
    }
    #pragma unroll
    for (int mi = 0; mi < 2; ++mi) {
        #pragma unroll
        for (int nj = 0; nj < 4; ++nj) {
            const int tm = m0 + wm * 32 + mi * 16;
            const int tn = n0 + wn * 64 + nj * 16;
            #pragma unroll
            for (int r = 0; r < 8; ++r) {
                const int gm = tm + lrow + r;
                const int gn = tn + lcol;
                const float v = acc[mi][nj][r];
                const size_t sidx = (size_t)gm * N + gn;
                size_t oidx = sidx;
                if constexpr (PERM_OUT) oidx = (size_t)permute_token(gm) * N + gn;
                if constexpr (EPI == EPI_NONE) {
                    C[oidx] = v;
                } else if constexpr (EPI == EPI_SILU) {
                    C[oidx] = siluf_(v);
                } else if constexpr (EPI == EPI_SILU2) {
                    C[oidx]  = siluf_(v);   // a1
                    C2[oidx] = v;           // raw h1 for silu'
                } else if constexpr (EPI == EPI_D2) {
                    float e = clip1_(v - aux[sidx]);          // err clip
                    C[oidx] = scaleArg * e;                   // d2
                } else if constexpr (EPI == EPI_D1) {
                    C[oidx] = v * silu_d_(aux[sidx]);         // d1
                } else {                                      // EPI_UPDATE
                    float g  = clip1_(v);                     // grad clip
                    float mm = Mbuf[sidx];
                    float ww = Wbuf[sidx];
                    float mn = g_et * mm - g_lr * g;
                    Mbuf[sidx] = mn;
                    Wbuf[sidx] = (1.0f - g_al) * ww + mn;
                }
            }
        }
    }
}

// ---- row-wise l2 normalize (one wave per 1024-wide row) -------------------
__global__ __launch_bounds__(256) void l2norm_kernel(float* __restrict__ buf) {
    const int lane = threadIdx.x & 31;
    const int wave = threadIdx.x >> 5;
    float* r = buf + ((size_t)blockIdx.x * 8 + wave) * DIMN;
    float ss = 0.0f;
    #pragma unroll
    for (int i = 0; i < DIMN / 128; ++i) {
        float4 v = *reinterpret_cast<const float4*>(r + i * 128 + lane * 4);
        ss += v.x * v.x + v.y * v.y + v.z * v.z + v.w * v.w;
    }
    #pragma unroll
    for (int off = 16; off > 0; off >>= 1) ss += __shfl_xor(ss, off, 32);
    const float inv = rsqrtf(ss + 1e-8f);
    #pragma unroll
    for (int i = 0; i < DIMN / 128; ++i) {
        float4 v = *reinterpret_cast<float4*>(r + i * 128 + lane * 4);
        v.x *= inv; v.y *= inv; v.z *= inv; v.w *= inv;
        *reinterpret_cast<float4*>(r + i * 128 + lane * 4) = v;
    }
}

// ---- per-chunk gate scalars: mean over (b, chunk-pos) of sigmoid(x.w + b) -
__global__ __launch_bounds__(256) void gates_kernel(
    const float* __restrict__ x,
    const float* __restrict__ Wlr,  const float* __restrict__ blr,
    const float* __restrict__ Wdec, const float* __restrict__ bdec,
    const float* __restrict__ Wmom, const float* __restrict__ bmom,
    float* __restrict__ gates)
{
    __shared__ float w[3][DIMN];
    __shared__ float red[3][8];
    const int t = threadIdx.x;
    const int c = blockIdx.x;
    for (int i = t; i < DIMN; i += 256) { w[0][i] = Wlr[i]; w[1][i] = Wdec[i]; w[2][i] = Wmom[i]; }
    __syncthreads();
    const int b = t >> 6, j = t & 63;
    const float* xr = x + ((size_t)b * SEQ + (size_t)c * CHUNKN + j) * DIMN;
    float s0 = 0.f, s1 = 0.f, s2 = 0.f;
    for (int i = 0; i < DIMN; i += 4) {
        float4 xv = *reinterpret_cast<const float4*>(xr + i);
        s0 += xv.x * w[0][i] + xv.y * w[0][i + 1] + xv.z * w[0][i + 2] + xv.w * w[0][i + 3];
        s1 += xv.x * w[1][i] + xv.y * w[1][i + 1] + xv.z * w[1][i + 2] + xv.w * w[1][i + 3];
        s2 += xv.x * w[2][i] + xv.y * w[2][i + 1] + xv.z * w[2][i + 2] + xv.w * w[2][i + 3];
    }
    float g0 = sigmoidf_(s0 + blr[0]);
    float g1 = sigmoidf_(s1 + bdec[0]);
    float g2 = sigmoidf_(s2 + bmom[0]);
    #pragma unroll
    for (int off = 16; off > 0; off >>= 1) {
        g0 += __shfl_xor(g0, off, 32);
        g1 += __shfl_xor(g1, off, 32);
        g2 += __shfl_xor(g2, off, 32);
    }
    const int lane = t & 31, wv = t >> 5;
    if (lane == 0) { red[0][wv] = g0; red[1][wv] = g1; red[2][wv] = g2; }
    __syncthreads();
    if (t == 0) {
        float a0 = 0.f, a1 = 0.f, a2 = 0.f;
        for (int i = 0; i < 8; ++i) { a0 += red[0][i]; a1 += red[1][i]; a2 += red[2][i]; }
        gates[c * 3 + 0] = a0 / 256.0f;   // lr
        gates[c * 3 + 1] = a1 / 256.0f;   // alpha
        gates[c * 3 + 2] = a2 / 256.0f;   // eta
    }
}

// ---------------------------------------------------------------------------
#define LAUNCH_GEMM(AT, BT, EPI, PERM, A, B, C, C2, AUX, W, MM, G, GI, M, N, K, SC)      \
    hipLaunchKernelGGL((wmma_gemm<AT, BT, EPI, PERM>), dim3((N) / TN, (M) / TM),         \
                       dim3(256), 0, stream, A, B, C, C2, AUX, W, MM, G, GI, M, N, K, SC)

extern "C" void kernel_launch(void* const* d_in, const int* in_sizes, int n_in,
                              void* d_out, int out_size, void* d_ws, size_t ws_size,
                              hipStream_t stream)
{
    const float* x    = (const float*)d_in[0];
    const float* Wk   = (const float*)d_in[1];
    const float* Wv   = (const float*)d_in[2];
    const float* Wq   = (const float*)d_in[3];
    const float* Wout = (const float*)d_in[4];
    const float* W1   = (const float*)d_in[5];
    const float* W2   = (const float*)d_in[6];
    const float* Wlr  = (const float*)d_in[7];
    const float* blr  = (const float*)d_in[8];
    const float* Wdec = (const float*)d_in[9];
    const float* bdec = (const float*)d_in[10];
    const float* Wmom = (const float*)d_in[11];
    const float* bmom = (const float*)d_in[12];
    float* out = (float*)d_out;

    const size_t TOKD = (size_t)TOKENS * DIMN;       // 8M floats
    const size_t WSZ  = (size_t)HIDN * DIMN;         // 2M floats
    float* ws  = (float*)d_ws;
    float* kc  = ws;  ws += TOKD;                    // chunk-permuted k
    float* vc  = ws;  ws += TOKD;                    // chunk-permuted v
    float* qc  = ws;  ws += TOKD;                    // chunk-permuted q
    float* yp  = ws;  ws += TOKD;                    // chunk-permuted y
    float* w1b = ws;  ws += WSZ;
    float* m1b = ws;  ws += WSZ;
    float* w2b = ws;  ws += WSZ;
    float* m2b = ws;  ws += WSZ;
    float* h1  = ws;  ws += (size_t)CROWS * HIDN;
    float* a1  = ws;  ws += (size_t)CROWS * HIDN;
    float* d1  = ws;  ws += (size_t)CROWS * HIDN;
    float* aq  = ws;  ws += (size_t)CROWS * HIDN;
    float* d2  = ws;  ws += (size_t)CROWS * DIMN;
    float* gts = ws;  ws += NCH * 3;

    const float scale = 2.0f / (float)(BSZ * CHUNKN * DIMN);

    // carry init (each call; graph-capturable async ops only)
    hipMemcpyAsync(w1b, W1, WSZ * sizeof(float), hipMemcpyDeviceToDevice, stream);
    hipMemcpyAsync(w2b, W2, WSZ * sizeof(float), hipMemcpyDeviceToDevice, stream);
    hipMemsetAsync(m1b, 0, WSZ * sizeof(float), stream);
    hipMemsetAsync(m2b, 0, WSZ * sizeof(float), stream);

    // ---- stage A: projections (+silu, write chunk-permuted), norms, gates -
    LAUNCH_GEMM(0, 0, EPI_SILU, true, x, Wk, kc, nullptr, nullptr, nullptr, nullptr,
                nullptr, 0, TOKENS, DIMN, DIMN, 0.f);
    LAUNCH_GEMM(0, 0, EPI_SILU, true, x, Wv, vc, nullptr, nullptr, nullptr, nullptr,
                nullptr, 0, TOKENS, DIMN, DIMN, 0.f);
    LAUNCH_GEMM(0, 0, EPI_SILU, true, x, Wq, qc, nullptr, nullptr, nullptr, nullptr,
                nullptr, 0, TOKENS, DIMN, DIMN, 0.f);
    hipLaunchKernelGGL(l2norm_kernel, dim3(TOKENS / 8), dim3(256), 0, stream, kc);
    hipLaunchKernelGGL(l2norm_kernel, dim3(TOKENS / 8), dim3(256), 0, stream, qc);
    hipLaunchKernelGGL(gates_kernel, dim3(NCH), dim3(256), 0, stream,
                       x, Wlr, blr, Wdec, bdec, Wmom, bmom, gts);

    // ---- stage B: sequential chunk scan ----------------------------------
    for (int c = 0; c < NCH; ++c) {
        const float* kcc = kc + (size_t)c * CROWS * DIMN;
        const float* vcc = vc + (size_t)c * CROWS * DIMN;
        const float* qcc = qc + (size_t)c * CROWS * DIMN;
        float*       ypc = yp + (size_t)c * CROWS * DIMN;
        // h1 = k.w1^T ; a1 = silu(h1)
        LAUNCH_GEMM(0, 0, EPI_SILU2, false, kcc, w1b, a1, h1, nullptr, nullptr, nullptr,
                    nullptr, 0, CROWS, HIDN, DIMN, 0.f);
        // d2 = scale * clip(a1.w2^T - v)
        LAUNCH_GEMM(0, 0, EPI_D2, false, a1, w2b, d2, nullptr, vcc, nullptr, nullptr,
                    nullptr, 0, CROWS, DIMN, HIDN, scale);
        // d1 = (d2.w2) * silu'(h1)   (reads OLD w2 -> must precede w2 update)
        LAUNCH_GEMM(0, 1, EPI_D1, false, d2, w2b, d1, nullptr, h1, nullptr, nullptr,
                    nullptr, 0, CROWS, HIDN, DIMN, 0.f);
        // g2 = clip(d2^T.a1); m2,w2 update in place
        LAUNCH_GEMM(1, 1, EPI_UPDATE, false, d2, a1, nullptr, nullptr, nullptr, w2b, m2b,
                    gts, c, DIMN, HIDN, CROWS, 0.f);
        // g1 = clip(d1^T.k); m1,w1 update in place
        LAUNCH_GEMM(1, 1, EPI_UPDATE, false, d1, kcc, nullptr, nullptr, nullptr, w1b, m1b,
                    gts, c, HIDN, DIMN, CROWS, 0.f);
        // retrieval with updated memory
        LAUNCH_GEMM(0, 0, EPI_SILU, false, qcc, w1b, aq, nullptr, nullptr, nullptr, nullptr,
                    nullptr, 0, CROWS, HIDN, DIMN, 0.f);
        LAUNCH_GEMM(0, 0, EPI_NONE, false, aq, w2b, ypc, nullptr, nullptr, nullptr, nullptr,
                    nullptr, 0, CROWS, DIMN, HIDN, 0.f);
    }

    // ---- stage C: out = y @ Wout^T (A-loader gathers permuted y rows) ----
    LAUNCH_GEMM(2, 0, EPI_NONE, false, yp, Wout, out, nullptr, nullptr, nullptr, nullptr,
                nullptr, 0, TOKENS, DIMN, DIMN, 0.f);
}
